// GraphNetwork_38843684225589
// MI455X (gfx1250) — matrix operand order repaired and checked by reading, not compiled, not run on hardware
//
#include <hip/hip_runtime.h>

#define NNODES   30000
#define NEDGES   480000
#define ETOT     (NNODES + NEDGES)   // 510000 edges incl. self loops
#define DMODEL   128
#define NHEADS   8
#define HDIM     16
#define NLAYERS  7
#define NEG_SLOPE 0.2f

typedef __attribute__((ext_vector_type(16))) _Float16 v16h;
typedef __attribute__((ext_vector_type(8)))  _Float16 v8h;
typedef __attribute__((ext_vector_type(8)))  float    v8f;

// ==================================================================
// Dense part: x@Wl+bl, x@Wr+br via v_wmma_f32_16x16x32_f16
// ==================================================================

// Convert x to f16 and pre-pack Wl/Wr into the WMMA B-fragment layout:
//   wp[((ks*8 + ntile)*32 + lane)*16 + h] = W[ks*32 + (lane<16?0:8) + h + (h<8?0:8), ntile*16 + (lane&15)]
__global__ void __launch_bounds__(256)
convert_pack_kernel(const float* __restrict__ x_in, _Float16* __restrict__ xf16,
                    const float* __restrict__ Wl, const float* __restrict__ Wr,
                    _Float16* __restrict__ wpL, _Float16* __restrict__ wpR)
{
    int t = blockIdx.x * blockDim.x + threadIdx.x;
    if (t < NNODES * DMODEL) xf16[t] = (_Float16)x_in[t];
    if (t < 4 * 8 * 32 * 16) {   // 16384 packed halves per matrix
        int h     = t & 15;
        int lane  = (t >> 4) & 31;
        int ntile = (t >> 9) & 7;
        int ks    = t >> 12;
        int koff  = (lane & 16) ? 8 : 0;
        int k = ks * 32 + koff + h + ((h >= 8) ? 8 : 0);
        int n = ntile * 16 + (lane & 15);
        wpL[t] = (_Float16)Wl[k * DMODEL + n];
        wpR[t] = (_Float16)Wr[k * DMODEL + n];
    }
}

// Block = 256 threads (8 wave32). Block owns 16 rows (LDS-staged once);
// wave w owns N-tile w, accumulating both xl and xr tiles (2 WMMA x 4 K-steps).
__global__ void __launch_bounds__(256)
gemm_xlxr_kernel(const _Float16* __restrict__ xf16,
                 const _Float16* __restrict__ wpL, const _Float16* __restrict__ wpR,
                 const float* __restrict__ bl, const float* __restrict__ br,
                 float* __restrict__ xl, float* __restrict__ xr)
{
    __shared__ _Float16 As[16 * DMODEL];     // 4 KB f16 strip

    const int lane = threadIdx.x & 31;
    const int wave = threadIdx.x >> 5;       // 0..7 -> N tile
    const int m0   = blockIdx.x << 4;        // 30000 = 1875 * 16, no tail

    {   // cooperative stage: 256 threads * 16 B = 4 KB
        int row = threadIdx.x >> 4;
        int c   = (threadIdx.x & 15) * 8;
        *(v8h*)(As + row * DMODEL + c) = *(const v8h*)(xf16 + (m0 + row) * DMODEL + c);
    }
    __syncthreads();

    // 16-bit A 16x32 layout: lane<16: M=lane, K{0..7,16..23}; lane>=16: M=lane-16, K{8..15,24..31}
    const int rl   = lane & 15;
    const int koff = (lane >> 4) << 3;
    const _Float16* aBase = As + rl * DMODEL + koff;

    v8f cl = {};
    v8f cr = {};
    #pragma unroll
    for (int ks = 0; ks < 4; ++ks) {
        const _Float16* ap = aBase + ks * 32;
        v8h alo = *(const v8h*)(ap);
        v8h ahi = *(const v8h*)(ap + 16);
        v16h a;
        #pragma unroll
        for (int i = 0; i < 8; ++i) { a[i] = alo[i]; a[i + 8] = ahi[i]; }

        const int bidx = (((ks << 3) | wave) * 32 + lane) << 4;
        v16h bL = *(const v16h*)(wpL + bidx);
        v16h bR = *(const v16h*)(wpR + bidx);

        cl = __builtin_amdgcn_wmma_f32_16x16x32_f16(false, a, false, bL, (short)0, cl, false, false);
        cr = __builtin_amdgcn_wmma_f32_16x16x32_f16(false, a, false, bR, (short)0, cr, false, false);
    }

    // C/D: VGPR v, lanes 0-15 -> M=v, lanes 16-31 -> M=v+8; N = lane&15
    const int row_add = (lane >> 4) << 3;
    const int col     = (wave << 4) + (lane & 15);
    const float bvl = bl[col];
    const float bvr = br[col];
    #pragma unroll
    for (int v = 0; v < 8; ++v) {
        int row = m0 + v + row_add;
        xl[row * DMODEL + col] = cl[v] + bvl;
        xr[row * DMODEL + col] = cr[v] + bvr;
    }
}

// ==================================================================
// CSR-by-destination build (once per call; dst is layer-invariant)
// ==================================================================

__device__ __forceinline__ int edge_dst(const int* __restrict__ ei, int e) {
    return (e < NEDGES) ? ei[NEDGES + e] : (e - NEDGES);
}
__device__ __forceinline__ int edge_src(const int* __restrict__ ei, int e) {
    return (e < NEDGES) ? ei[e] : (e - NEDGES);
}

__global__ void __launch_bounds__(256)
zero_deg_kernel(int* __restrict__ deg)
{
    int t = blockIdx.x * blockDim.x + threadIdx.x;
    if (t < NNODES) deg[t] = 0;
}

__global__ void __launch_bounds__(256)
count_deg_kernel(const int* __restrict__ ei, int* __restrict__ deg)
{
    int e = blockIdx.x * blockDim.x + threadIdx.x;
    if (e >= ETOT) return;
    atomicAdd(&deg[edge_dst(ei, e)], 1);
}

// Single-block exclusive scan over deg[0..NNODES) -> rowptr[0..NNODES], cursor copy.
__global__ void __launch_bounds__(1024)
scan_kernel(const int* __restrict__ deg, int* __restrict__ rowptr, int* __restrict__ cursor)
{
    __shared__ int sums[1024];
    const int t  = threadIdx.x;
    const int CH = (NNODES + 1023) / 1024;   // 30
    const int base = t * CH;

    int local = 0;
    for (int i = 0; i < CH; ++i) {
        int idx = base + i;
        if (idx < NNODES) local += deg[idx];
    }
    sums[t] = local;
    __syncthreads();
    // Hillis-Steele inclusive scan
    for (int s = 1; s < 1024; s <<= 1) {
        int add = (t >= s) ? sums[t - s] : 0;
        __syncthreads();
        sums[t] += add;
        __syncthreads();
    }
    int run = sums[t] - local;               // exclusive prefix of this chunk
    for (int i = 0; i < CH; ++i) {
        int idx = base + i;
        if (idx < NNODES) {
            rowptr[idx] = run;
            cursor[idx] = run;
            run += deg[idx];
        }
    }
    if (t == 1023) rowptr[NNODES] = sums[1023];
}

__global__ void __launch_bounds__(256)
scatter_kernel(const int* __restrict__ ei, int* __restrict__ cursor, int* __restrict__ eid)
{
    int e = blockIdx.x * blockDim.x + threadIdx.x;
    if (e >= ETOT) return;
    int pos = atomicAdd(&cursor[edge_dst(ei, e)], 1);
    eid[pos] = e;
}

// ==================================================================
// Fused per-node GATv2 attention: one wave32 per destination node.
// Lane l owns channels 4l..4l+3 (head h = l>>2). Exact segment softmax,
// zero atomics, out = bias + sum(alpha * xl[src]).
// ==================================================================
__global__ void __launch_bounds__(256)
node_attn_kernel(const float* __restrict__ xl, const float* __restrict__ xr,
                 const float* __restrict__ att, const int* __restrict__ ei,
                 const int* __restrict__ rowptr, const int* __restrict__ eid,
                 const float* __restrict__ bias, float* __restrict__ evals,
                 float* __restrict__ xout)
{
    const int lane = threadIdx.x & 31;
    const int n = blockIdx.x * (blockDim.x >> 5) + (threadIdx.x >> 5);  // 3750*8 = 30000 exact
    const int h  = lane >> 2;
    const int c0 = lane << 2;

    float4 xrv = *(const float4*)(xr + (size_t)n * DMODEL + c0);
    float4 atv = *(const float4*)(att + h * HDIM + (lane & 3) * 4);

    const int beg = rowptr[n];
    const int end = rowptr[n + 1];   // >= beg+1 (self loop guarantees non-empty)

    // ---- pass 1: scores + running max ----
    float hmax = -3.402823466e+38f;
    for (int pos = beg; pos < end; ++pos) {
        int e = eid[pos];
        int s = edge_src(ei, e);
        float4 xlv = *(const float4*)(xl + (size_t)s * DMODEL + c0);
        float m0 = xlv.x + xrv.x; m0 = (m0 > 0.f) ? m0 : NEG_SLOPE * m0;
        float m1 = xlv.y + xrv.y; m1 = (m1 > 0.f) ? m1 : NEG_SLOPE * m1;
        float m2 = xlv.z + xrv.z; m2 = (m2 > 0.f) ? m2 : NEG_SLOPE * m2;
        float m3 = xlv.w + xrv.w; m3 = (m3 > 0.f) ? m3 : NEG_SLOPE * m3;
        float p = m0 * atv.x + m1 * atv.y + m2 * atv.z + m3 * atv.w;
        // butterfly reduce across the 4 lanes of this head -> all 4 hold total
        p += __shfl_xor(p, 1, 32);
        p += __shfl_xor(p, 2, 32);
        if ((lane & 3) == 0) evals[(size_t)pos * NHEADS + h] = p;
        hmax = fmaxf(hmax, p);
    }

    // ---- pass 2: exp + denom ----
    float dsum = 0.f;
    for (int pos = beg; pos < end; ++pos) {
        float sc = evals[(size_t)pos * NHEADS + h];
        float ee = __expf(sc - hmax);
        if ((lane & 3) == 0) evals[(size_t)pos * NHEADS + h] = ee;
        dsum += ee;
    }
    const float inv = 1.f / (dsum + 1e-16f);

    // ---- pass 3: weighted aggregation ----
    float a0 = 0.f, a1 = 0.f, a2 = 0.f, a3 = 0.f;
    for (int pos = beg; pos < end; ++pos) {
        int e = eid[pos];
        int s = edge_src(ei, e);
        float alpha = evals[(size_t)pos * NHEADS + h] * inv;
        float4 xlv = *(const float4*)(xl + (size_t)s * DMODEL + c0);
        a0 += alpha * xlv.x;
        a1 += alpha * xlv.y;
        a2 += alpha * xlv.z;
        a3 += alpha * xlv.w;
    }
    float4 bv = *(const float4*)(bias + c0);
    float4 outv = make_float4(a0 + bv.x, a1 + bv.y, a2 + bv.z, a3 + bv.w);
    *(float4*)(xout + (size_t)n * DMODEL + c0) = outv;
}

// ==================================================================
// y[0:2] = (x @ W_out + b_out)[0:2]
// ==================================================================
__global__ void head_out_kernel(const float* __restrict__ x, const float* __restrict__ Wout,
                                const float* __restrict__ bout, float* __restrict__ y)
{
    int i = threadIdx.x;
    if (i >= 2) return;
    float acc = 0.0f;
    #pragma unroll 8
    for (int k = 0; k < DMODEL; ++k) acc += x[i * DMODEL + k] * Wout[k];
    y[i] = acc + bout[0];
}

extern "C" void kernel_launch(void* const* d_in, const int* in_sizes, int n_in,
                              void* d_out, int out_size, void* d_ws, size_t ws_size,
                              hipStream_t stream)
{
    (void)in_sizes; (void)n_in; (void)out_size; (void)ws_size;

    const float* x0   = (const float*)d_in[0];
    const int*   ei   = (const int*)  d_in[1];   // edge_index flattened [2,E]
    const float* Wl   = (const float*)d_in[2];   // [7,128,128]
    const float* bl   = (const float*)d_in[3];   // [7,128]
    const float* Wr   = (const float*)d_in[4];
    const float* br   = (const float*)d_in[5];
    const float* att  = (const float*)d_in[6];   // [7,8,16]
    const float* bias = (const float*)d_in[7];   // [7,128]
    const float* Wout = (const float*)d_in[8];   // [128,1]
    const float* bout = (const float*)d_in[9];   // [1]
    float* out = (float*)d_out;

    // ---- workspace carve-out (256B aligned) ----
    char* ws = (char*)d_ws;
    size_t off = 0;
    auto carve = [&](size_t bytes) -> char* {
        char* p = ws + off;
        off += (bytes + 255) & ~(size_t)255;
        return p;
    };
    float*    bufA   = (float*)   carve((size_t)NNODES * DMODEL * 4);
    float*    bufB   = (float*)   carve((size_t)NNODES * DMODEL * 4);
    float*    xlb    = (float*)   carve((size_t)NNODES * DMODEL * 4);
    float*    xrb    = (float*)   carve((size_t)NNODES * DMODEL * 4);
    _Float16* xf16   = (_Float16*)carve((size_t)NNODES * DMODEL * 2);
    _Float16* wpL    = (_Float16*)carve((size_t)16384 * 2);
    _Float16* wpR    = (_Float16*)carve((size_t)16384 * 2);
    float*    evals  = (float*)   carve((size_t)ETOT * NHEADS * 4);
    int*      deg    = (int*)     carve((size_t)NNODES * 4);
    int*      cursor = (int*)     carve((size_t)NNODES * 4);
    int*      rowptr = (int*)     carve((size_t)(NNODES + 1) * 4);
    int*      eid    = (int*)     carve((size_t)ETOT * 4);

    const int gNode  = (NNODES * DMODEL + 255) / 256;
    const int gEdgeE = (ETOT + 255) / 256;

    // ---- build CSR by destination (dst is identical for all layers) ----
    zero_deg_kernel<<<(NNODES + 255) / 256, 256, 0, stream>>>(deg);
    count_deg_kernel<<<gEdgeE, 256, 0, stream>>>(ei, deg);
    scan_kernel<<<1, 1024, 0, stream>>>(deg, rowptr, cursor);
    scatter_kernel<<<gEdgeE, 256, 0, stream>>>(ei, cursor, eid);

    for (int L = 0; L < NLAYERS; ++L) {
        const float* xin  = (L == 0) ? x0 : ((L & 1) ? bufA : bufB);
        float*       xout = (L & 1) ? bufB : bufA;

        convert_pack_kernel<<<gNode, 256, 0, stream>>>(
            xin, xf16, Wl + (size_t)L * DMODEL * DMODEL, Wr + (size_t)L * DMODEL * DMODEL,
            wpL, wpR);

        gemm_xlxr_kernel<<<NNODES / 16, 256, 0, stream>>>(
            xf16, wpL, wpR, bl + L * DMODEL, br + L * DMODEL, xlb, xrb);

        node_attn_kernel<<<NNODES / 8, 256, 0, stream>>>(
            xlb, xrb, att + L * NHEADS * HDIM, ei, rowptr, eid,
            bias + L * DMODEL, evals, xout);
    }

    // 7 layers: final x lives in bufA (layers 0,2,4,6 write bufA)
    head_out_kernel<<<1, 32, 0, stream>>>(bufA, Wout, bout, out);
}
